// HNetLM_89352499626295
// MI455X (gfx1250) — matrix operand order repaired and checked by reading, not compiled, not run on hardware
//
#include <hip/hip_runtime.h>
#include <stddef.h>
#include <stdint.h>

// ---------------------------------------------------------------------------
// HNet forward on MI455X (gfx1250, wave32).
//
// Dominant cost: 3 GEMMs of (B*L=16384, D=1024) x (1024,1024) = ~103 GFLOP.
// Done with v_wmma_f32_16x16x32_bf16 using a 3-term bf16 split
// (hi*hi + hi*lo + lo*hi, fp32 accumulate) for near-fp32 accuracy on the
// bf16 matrix pipe. Wave tile 32x64 -> 24 WMMA per 24 b128 loads per K-step,
// keeping the XDL pipe fed from the 192MB L2 (X bf16 + weights fully
// resident). All other stages are bandwidth-bound elementwise / scan work
// (~300MB total traffic -> ~13us at 23.3 TB/s HBM).
// ---------------------------------------------------------------------------

typedef __attribute__((ext_vector_type(16))) __bf16 v16bf;
typedef __attribute__((ext_vector_type(8)))  float  v8f;

union FragU { uint4 u[2]; v16bf v; };

// ----------------------------- bf16 split ----------------------------------
__device__ __forceinline__ unsigned short f32_to_bf16_rne(float f) {
    unsigned u = __float_as_uint(f);
    u += 0x7FFFu + ((u >> 16) & 1u);   // round-to-nearest-even (finite inputs)
    return (unsigned short)(u >> 16);
}

__global__ void split_bf16_kernel(const float* __restrict__ src,
                                  unsigned short* __restrict__ hi,
                                  unsigned short* __restrict__ lo,
                                  size_t n) {
    size_t i = (size_t)blockIdx.x * blockDim.x + threadIdx.x;
    if (i >= n) return;
    float f = src[i];
    unsigned short h = f32_to_bf16_rne(f);
    float hf = __uint_as_float((unsigned)h << 16);
    unsigned short l = f32_to_bf16_rne(f - hf);
    hi[i] = h;
    lo[i] = l;
}

// ----------------------------- WMMA GEMM -----------------------------------
// A fragment: 16x32 bf16 (MxK). Lane = (row M = lane&15, kgroup = lane>>4).
// Per ISA: v0..3 lanes0-15 hold K=0..7 (2/vgpr), lanes16-31 hold K=8..15,
// v4..7 hold K=16..31 likewise -> two contiguous 16B runs per lane.
__device__ __forceinline__ v16bf load_a_frag(const unsigned short* __restrict__ X,
                                             int ld, int m0, int k0, int lane) {
    int row = lane & 15, kg = lane >> 4;
    const unsigned short* r = X + (size_t)(m0 + row) * ld + k0 + kg * 8;
    FragU f;
    f.u[0] = *(const uint4*)(r);        // K = kg*8 .. kg*8+7
    f.u[1] = *(const uint4*)(r + 16);   // K = 16+kg*8 .. +7
    __builtin_prefetch((const void*)(r + 64), 0, 3); // two K-tiles ahead, WGP scope
    return f.v;
}

// B fragment: 32x16 bf16 (KxN) from row-major W[n,k] (i.e. B = W^T).
// Lane = (col N = lane&15, kgroup = lane>>4): same K order as the A fragment.
__device__ __forceinline__ v16bf load_b_frag(const unsigned short* __restrict__ W,
                                             int ld, int n0, int k0, int lane) {
    int col = lane & 15, kh = lane >> 4;
    const unsigned short* r = W + (size_t)(n0 + col) * ld + k0 + kh * 8;
    FragU f;
    f.u[0] = *(const uint4*)(r);
    f.u[1] = *(const uint4*)(r + 16);
    __builtin_prefetch((const void*)(r + 64), 0, 3);
    return f.v;
}

__device__ __forceinline__ v8f wmma_bf(v16bf a, v16bf b, v8f c) {
    return __builtin_amdgcn_wmma_f32_16x16x32_bf16(
        /*neg_a=*/false, a, /*neg_b=*/false, b,
        /*c_mod=*/(short)0, c, /*reuse_a=*/false, /*reuse_b=*/false);
}

// out[m,n] = sum_k X[m,k] * W[n,k]  (+ bias[n]); 3-term bf16 compensation.
// Block: 256 threads = 8 waves (2 in m, 4 in n); block tile 64(m) x 256(n);
// wave tile 32(m) x 64(n) -> 8 f32 accumulators, 24 WMMA per K-step.
__global__ __launch_bounds__(256) void gemm_bf16x3_kernel(
    const unsigned short* __restrict__ Xh, const unsigned short* __restrict__ Xl,
    const unsigned short* __restrict__ Wh, const unsigned short* __restrict__ Wl,
    const float* __restrict__ bias, float* __restrict__ out, int K, int N) {
    const int lane = threadIdx.x & 31;
    const int wave = threadIdx.x >> 5;
    const int wm = wave & 1, wn = wave >> 1;
    const int m0 = blockIdx.y * 64 + wm * 32;
    const int n0 = blockIdx.x * 256 + wn * 64;

    v8f acc[2][4];
#pragma unroll
    for (int i = 0; i < 2; ++i)
#pragma unroll
        for (int j = 0; j < 4; ++j) acc[i][j] = (v8f){};

    for (int k0 = 0; k0 < K; k0 += 32) {
        v16bf ah0 = load_a_frag(Xh, K, m0,      k0, lane);
        v16bf ah1 = load_a_frag(Xh, K, m0 + 16, k0, lane);
        v16bf al0 = load_a_frag(Xl, K, m0,      k0, lane);
        v16bf al1 = load_a_frag(Xl, K, m0 + 16, k0, lane);
        v16bf bh[4], bl[4];
#pragma unroll
        for (int j = 0; j < 4; ++j) {
            bh[j] = load_b_frag(Wh, K, n0 + j * 16, k0, lane);
            bl[j] = load_b_frag(Wl, K, n0 + j * 16, k0, lane);
        }
#pragma unroll
        for (int j = 0; j < 4; ++j) {
            // hi*hi
            acc[0][j] = wmma_bf(ah0, bh[j], acc[0][j]);
            acc[1][j] = wmma_bf(ah1, bh[j], acc[1][j]);
            // hi*lo
            acc[0][j] = wmma_bf(ah0, bl[j], acc[0][j]);
            acc[1][j] = wmma_bf(ah1, bl[j], acc[1][j]);
            // lo*hi
            acc[0][j] = wmma_bf(al0, bh[j], acc[0][j]);
            acc[1][j] = wmma_bf(al1, bh[j], acc[1][j]);
        }
    }

    // C/D layout: lane col N = lane&15; VGPR r holds row M = r + 8*(lane>>4).
    const int col  = lane & 15;
    const int rsel = (lane >> 4) * 8;
#pragma unroll
    for (int i = 0; i < 2; ++i) {
#pragma unroll
        for (int j = 0; j < 4; ++j) {
            int n = n0 + j * 16 + col;
            float bv = bias ? bias[n] : 0.0f;
#pragma unroll
            for (int r = 0; r < 8; ++r) {
                int m = m0 + i * 16 + rsel + r;
                out[(size_t)m * N + n] = acc[i][j][r] + bv;
            }
        }
    }
}

// ----------------------------- router --------------------------------------
// prob[b,0]=1; prob[b,t] = clip((1 - cos(norm(qv[t-1]), norm(kv[t])))/2, 0, 1)
__global__ __launch_bounds__(128) void router_kernel(
    const float* __restrict__ QV, const float* __restrict__ KV,
    float* __restrict__ prob, int L, int D) {
    const int idx = blockIdx.x;          // b*L + t
    const int t = idx % L;
    const int tid = threadIdx.x;
    if (t == 0) {
        if (tid == 0) prob[idx] = 1.0f;
        return;
    }
    const float* q = QV + (size_t)(idx - 1) * D;  // same batch since t >= 1
    const float* k = KV + (size_t)idx * D;
    float sqk = 0.f, sqq = 0.f, skk = 0.f;
    for (int d = tid; d < D; d += 128) {
        float qa = q[d], ka = k[d];
        sqk += qa * ka; sqq += qa * qa; skk += ka * ka;
    }
    __shared__ float s0[128], s1[128], s2[128];
    s0[tid] = sqk; s1[tid] = sqq; s2[tid] = skk;
    __syncthreads();
    for (int off = 64; off > 0; off >>= 1) {
        if (tid < off) {
            s0[tid] += s0[tid + off];
            s1[tid] += s1[tid + off];
            s2[tid] += s2[tid + off];
        }
        __syncthreads();
    }
    if (tid == 0) {
        float nq = fmaxf(sqrtf(s1[0]), 1e-12f);
        float nk = fmaxf(sqrtf(s2[0]), 1e-12f);
        float cs = s0[0] / (nq * nk);
        prob[idx] = fminf(fmaxf((1.0f - cs) * 0.5f, 0.0f), 1.0f);
    }
}

// ------------------------ compaction / prefix scan -------------------------
// One block per batch; 256 threads, each owning L/256 (<=32) positions.
__global__ __launch_bounds__(256) void scan_kernel(
    const float* __restrict__ prob, int* __restrict__ pos,
    float* __restrict__ cprob, int* __restrict__ chunk_idx,
    int* __restrict__ count, int L) {
    const int b = blockIdx.x;
    const int tid = threadIdx.x;
    const int per = L / 256;             // 16 for L=4096
    const size_t base = (size_t)b * L;
    __shared__ int s[256];

    unsigned mbits = 0u;
    int local = 0;
    for (int i = 0; i < per; ++i) {
        float p = prob[base + tid * per + i];
        int m = p > 0.5f;
        mbits |= ((unsigned)m << i);
        local += m;
    }
    s[tid] = local;
    __syncthreads();
    for (int off = 1; off < 256; off <<= 1) {   // Hillis-Steele inclusive scan
        int v = s[tid];
        int add = (tid >= off) ? s[tid - off] : 0;
        __syncthreads();
        s[tid] = v + add;
        __syncthreads();
    }
    int run = (tid == 0) ? 0 : s[tid - 1];      // exclusive prefix
    for (int i = 0; i < per; ++i) {
        int l = tid * per + i;
        if ((mbits >> i) & 1u) {
            pos[base + run]   = l;
            cprob[base + run] = prob[base + l];
            ++run;
        }
        chunk_idx[base + l] = run - 1;          // cumsum(mask) - 1
    }
    if (tid == 255) count[b] = s[255];
}

// ----------------------------- EMA scan ------------------------------------
// h[j] = decay[j]*h[j-1] + (1-decay[j])*x[pos[j]], decay = clip(1-p, 0, 1).
// One block per batch, one thread per channel d; per-step loads coalesced.
__global__ __launch_bounds__(1024) void ema_kernel(
    const float* __restrict__ x, const float* __restrict__ init_state,
    const int* __restrict__ pos, const float* __restrict__ cprob,
    const int* __restrict__ count, float* __restrict__ ema, int L, int D) {
    const int b = blockIdx.x;
    const int d = threadIdx.x;
    const size_t baseBL = (size_t)b * L;
    float h = init_state[(size_t)b * D + d];
    const int cnt = count[b];
    for (int j = 0; j < cnt; ++j) {
        float p = cprob[baseBL + j];
        float a = fminf(fmaxf(1.0f - p, 0.0f), 1.0f);
        int l = pos[baseBL + j];
        float xv = x[(baseBL + l) * (size_t)D + d];
        h = a * h + (1.0f - a) * xv;
        ema[(baseBL + j) * (size_t)D + d] = h;
    }
}

// ----------------------------- upsample + add ------------------------------
// out[b,l,:] = residual(already in out) + ema[b, chunk_idx[b,l], :]  (coef==1)
__global__ __launch_bounds__(256) void finalize_kernel(
    const float* __restrict__ ema, const int* __restrict__ chunk_idx,
    float* __restrict__ out, int L, int D) {
    const int idx = blockIdx.x;          // b*L + l
    const int ci = chunk_idx[idx];
    if (ci < 0) return;
    const int b = idx / L;
    const float4* src = (const float4*)(ema + ((size_t)b * L + ci) * D);
    float4* dst = (float4*)(out + (size_t)idx * D);
    const int t = threadIdx.x;           // D/4 threads
    float4 o = dst[t];
    float4 s = src[t];
    o.x += s.x; o.y += s.y; o.z += s.z; o.w += s.w;
    dst[t] = o;
}

// ----------------------------- launcher ------------------------------------
extern "C" void kernel_launch(void* const* d_in, const int* in_sizes, int n_in,
                              void* d_out, int out_size, void* d_ws, size_t ws_size,
                              hipStream_t stream) {
    const float* x     = (const float*)d_in[0];
    const float* Wq    = (const float*)d_in[1];
    const float* Wk    = (const float*)d_in[2];
    const float* Wr    = (const float*)d_in[3];
    const float* br    = (const float*)d_in[4];
    const float* inits = (const float*)d_in[5];

    const int    D = in_sizes[4];                    // 1024
    const int    B = in_sizes[5] / D;                // 4
    const int    L = (int)((size_t)in_sizes[0] / ((size_t)B * D)); // 4096
    const size_t M = (size_t)B * L;                  // 16384 rows

    // ---- workspace layout (256B-aligned slices) ----
    char* wsp = (char*)d_ws;
    auto alloc = [&](size_t bytes) -> void* {
        void* p = (void*)wsp;
        wsp += (bytes + 255) & ~(size_t)255;
        return p;
    };
    unsigned short* Xh  = (unsigned short*)alloc(M * D * sizeof(unsigned short));
    unsigned short* Xl  = (unsigned short*)alloc(M * D * sizeof(unsigned short));
    unsigned short* Wqh = (unsigned short*)alloc((size_t)D * D * sizeof(unsigned short));
    unsigned short* Wql = (unsigned short*)alloc((size_t)D * D * sizeof(unsigned short));
    unsigned short* Wkh = (unsigned short*)alloc((size_t)D * D * sizeof(unsigned short));
    unsigned short* Wkl = (unsigned short*)alloc((size_t)D * D * sizeof(unsigned short));
    unsigned short* Wrh = (unsigned short*)alloc((size_t)D * D * sizeof(unsigned short));
    unsigned short* Wrl = (unsigned short*)alloc((size_t)D * D * sizeof(unsigned short));
    float* QV    = (float*)alloc(M * D * sizeof(float));  // reused as `ema` later
    float* KV    = (float*)alloc(M * D * sizeof(float));
    float* prob  = (float*)alloc(M * sizeof(float));
    int*   pos   = (int*)alloc(M * sizeof(int));
    float* cprob = (float*)alloc(M * sizeof(float));
    int*   cidx  = (int*)alloc(M * sizeof(int));
    int*   count = (int*)alloc((size_t)B * sizeof(int));
    float* ema   = QV;                                   // QV dead after router

    float* out = (float*)d_out;

    // 1) bf16 hi/lo splits of X and the three weight matrices
    {
        size_t nx = M * (size_t)D;
        split_bf16_kernel<<<dim3((unsigned)((nx + 255) / 256)), 256, 0, stream>>>(x, Xh, Xl, nx);
        size_t nw = (size_t)D * D;
        unsigned gw = (unsigned)((nw + 255) / 256);
        split_bf16_kernel<<<gw, 256, 0, stream>>>(Wq, Wqh, Wql, nw);
        split_bf16_kernel<<<gw, 256, 0, stream>>>(Wk, Wkh, Wkl, nw);
        split_bf16_kernel<<<gw, 256, 0, stream>>>(Wr, Wrh, Wrl, nw);
    }

    // 2) three WMMA GEMMs: QV = x@Wq^T, KV = x@Wk^T, residual = x@Wr^T + br
    {
        dim3 grid((unsigned)(D / 256), (unsigned)(M / 64));
        gemm_bf16x3_kernel<<<grid, 256, 0, stream>>>(Xh, Xl, Wqh, Wql, nullptr, QV, D, D);
        gemm_bf16x3_kernel<<<grid, 256, 0, stream>>>(Xh, Xl, Wkh, Wkl, nullptr, KV, D, D);
        gemm_bf16x3_kernel<<<grid, 256, 0, stream>>>(Xh, Xl, Wrh, Wrl, br, out, D, D);
    }

    // 3) boundary probabilities
    router_kernel<<<dim3((unsigned)M), 128, 0, stream>>>(QV, KV, prob, L, D);

    // 4) per-batch compaction + chunk indices
    scan_kernel<<<dim3((unsigned)B), 256, 0, stream>>>(prob, pos, cprob, cidx, count, L);

    // 5) EMA scan over compacted chunks (ema aliases QV scratch)
    ema_kernel<<<dim3((unsigned)B), (unsigned)D, 0, stream>>>(x, inits, pos, cprob, count, ema, L, D);

    // 6) ragged upsample + residual add into d_out
    finalize_kernel<<<dim3((unsigned)M), (unsigned)(D / 4), 0, stream>>>(ema, cidx, out, L, D);
}
